// QFNetBlock_52828097740806
// MI455X (gfx1250) — matrix operand order repaired and checked by reading, not compiled, not run on hardware
//
#include <hip/hip_runtime.h>
#include <math.h>

#define DIM 4096
#define NB  128

typedef __attribute__((ext_vector_type(16))) _Float16 v16h;
typedef __attribute__((ext_vector_type(8)))  _Float16 v8h;
typedef __attribute__((ext_vector_type(4)))  _Float16 v4h;
typedef __attribute__((ext_vector_type(8)))  float    v8f;

// ---------------------------------------------------------------------------
// Stage 1: L2-normalize each batch row (StateEncoder). imag part stays 0.
// ---------------------------------------------------------------------------
__global__ __launch_bounds__(256) void encode_kernel(const float* __restrict__ x,
                                                     float* __restrict__ outRe) {
  __shared__ float red[256];
  const int b = blockIdx.x;
  const float* row = x + (size_t)b * DIM;
  float s = 0.f;
  for (int i = threadIdx.x; i < DIM; i += 256) { const float v = row[i]; s += v * v; }
  red[threadIdx.x] = s;
  __syncthreads();
  for (int off = 128; off > 0; off >>= 1) {
    if (threadIdx.x < off) red[threadIdx.x] += red[threadIdx.x + off];
    __syncthreads();
  }
  const float inv = rsqrtf(red[0]);
  float* o = outRe + (size_t)b * DIM;
  for (int i = threadIdx.x; i < DIM; i += 256) o[i] = row[i] * inv;
}

// ---------------------------------------------------------------------------
// CNOT ring: composed basis permutation.  state_out[y] = state_in[P(y)] where
// P(y) applies gate 11, 10, ..., 0; gate i: control wire i (bit 11-i),
// target wire (i+1)%12 (bit 11-((i+1)%12)), flip target if control set.
// ---------------------------------------------------------------------------
__global__ __launch_bounds__(256) void cnot_kernel(const float* __restrict__ inRe,
                                                   const float* __restrict__ inIm,
                                                   float* __restrict__ outRe,
                                                   float* __restrict__ outIm) {
  const int idx = blockIdx.x * 256 + threadIdx.x;
  const int b = idx >> 12;
  const int y = idx & (DIM - 1);
  int s = y;
#pragma unroll
  for (int i = 11; i >= 0; --i) {
    const int c = 11 - i;
    const int t = 11 - ((i + 1) % 12);
    s ^= ((s >> c) & 1) << t;
  }
  outRe[(size_t)b * DIM + y] = inRe[(size_t)b * DIM + s];
  outIm[(size_t)b * DIM + y] = inIm[(size_t)b * DIM + s];
}

// ---------------------------------------------------------------------------
// QFT: one workgroup per batch element, statevector resident in 32 KB LDS.
// For m = 11..1: H butterfly on bit p = 11-m, then the fused diagonal of all
// CP(pi/2^(m-i)) gates: theta(x) = pi * bitrev(top m bits of x) / 2^m, applied
// where bit p is set.  SWAP network at the end == 12-bit reversal permutation.
// ---------------------------------------------------------------------------
__global__ __launch_bounds__(256) void qft_kernel(const float* __restrict__ inRe,
                                                  const float* __restrict__ inIm,
                                                  float* __restrict__ outRe,
                                                  float* __restrict__ outIm) {
  __shared__ float qre[DIM];
  __shared__ float qim[DIM];
  const int b = blockIdx.x;
  const size_t base = (size_t)b * DIM;
  for (int i = threadIdx.x; i < DIM; i += 256) { qre[i] = inRe[base + i]; qim[i] = inIm[base + i]; }
  __syncthreads();
  const float RS2 = 0.70710678118654752f;
  for (int m = 11; m >= 1; --m) {
    const int p = 11 - m;
    // Hadamard on wire m (bit p)
    for (int t = threadIdx.x; t < DIM / 2; t += 256) {
      const int lo = ((t >> p) << (p + 1)) | (t & ((1 << p) - 1));
      const int hi = lo | (1 << p);
      const float ar = qre[lo], ai = qim[lo], br = qre[hi], bi = qim[hi];
      qre[lo] = (ar + br) * RS2; qim[lo] = (ai + bi) * RS2;
      qre[hi] = (ar - br) * RS2; qim[hi] = (ai - bi) * RS2;
    }
    __syncthreads();
    // fused controlled-phase diagonal
    for (int xq = threadIdx.x; xq < DIM; xq += 256) {
      if ((xq >> p) & 1) {
        const unsigned rev = __brev((unsigned)xq >> (12 - m)) >> (32 - m);
        const float theta = 3.14159265358979f * (float)rev / (float)(1 << m);
        const float cs = cosf(theta), sn = sinf(theta);
        const float r0 = qre[xq], i0 = qim[xq];
        qre[xq] = r0 * cs - i0 * sn;
        qim[xq] = r0 * sn + i0 * cs;
      }
    }
    __syncthreads();
  }
  for (int i = threadIdx.x; i < DIM; i += 256) {
    const int r = __brev((unsigned)i) >> 20;  // 12-bit reversal (SWAP network)
    outRe[base + i] = qre[r];
    outIm[base + i] = qim[r];
  }
}

// ---------------------------------------------------------------------------
// Complex GEMM: out[b,i] = sum_j U[i,j] * s[b,j].
// D(M=4096 x N=128) = U(MxK) * S^T(KxN), K = 4096.
// 128 threads = 4 waves (2x2); wave computes 32x32 complex output (2x2 WMMA
// tiles).  K-step 32, double-buffered f16 LDS staging: next tile's
// global_load_b128s are issued before the current tile's WMMA block.
// ---------------------------------------------------------------------------
__device__ __forceinline__ void load_regs(const float* __restrict__ src, int row0, int k0,
                                          int tid, v4h out[4]) {
#pragma unroll
  for (int t = 0; t < 4; ++t) {
    const int idx = tid + t * 128;   // 0..511 quads over 64x32 tile
    const int r = idx >> 3;          // 0..63
    const int c = (idx & 7) * 4;     // 0..28
    const float4 v = *(const float4*)(src + (size_t)(row0 + r) * DIM + k0 + c);
    v4h h;
    h.x = (_Float16)v.x; h.y = (_Float16)v.y; h.z = (_Float16)v.z; h.w = (_Float16)v.w;
    out[t] = h;
  }
}

__device__ __forceinline__ void store_regs(const v4h in[4], _Float16* __restrict__ dst, int tid) {
#pragma unroll
  for (int t = 0; t < 4; ++t) {
    const int idx = tid + t * 128;
    const int r = idx >> 3;
    const int c = (idx & 7) * 4;
    *(v4h*)(dst + r * 32 + c) = in[t];
  }
}

// A fragment (16x32 f16, ISA layout): lanes 0-15 row M=lane, K {0..7,16..23};
// lanes 16-31 row M=lane-16, K {8..15,24..31}.  Two ds_load_b128 per lane.
__device__ __forceinline__ v16h load_frag_a(const _Float16* lds, int lane, int rowbase) {
  const int r = rowbase + (lane & 15);
  const int kh = (lane >> 4) * 8;
  const v8h lo = *(const v8h*)(lds + r * 32 + kh);
  const v8h hi = *(const v8h*)(lds + r * 32 + kh + 16);
  return __builtin_shufflevector(lo, hi, 0, 1, 2, 3, 4, 5, 6, 7, 8, 9, 10, 11, 12, 13, 14, 15);
}

// B fragment (32x16 f16): lanes 0-15 col N=lane K {0..7,16..23}; lanes 16-31 col
// N=lane-16 K {8..15,24..31}.  Matches the A fragment's K order.
__device__ __forceinline__ v16h load_frag_b(const _Float16* lds, int lane, int rowbase) {
  const int r = rowbase + (lane & 15);
  const int kb = (lane >> 4) * 8;
  const v8h lo = *(const v8h*)(lds + r * 32 + kb);
  const v8h hi = *(const v8h*)(lds + r * 32 + kb + 16);
  return __builtin_shufflevector(lo, hi, 0, 1, 2, 3, 4, 5, 6, 7, 8, 9, 10, 11, 12, 13, 14, 15);
}

__device__ __forceinline__ v8f wmma(v16h a, v16h b, v8f c) {
  return __builtin_amdgcn_wmma_f32_16x16x32_f16(false, a, false, b, (short)0, c, false, false);
}

template <bool HAS_IM, bool ABS>
__global__ __launch_bounds__(128) void cgemm_kernel(
    const float* __restrict__ Ure, const float* __restrict__ Uim,
    const float* __restrict__ inRe, const float* __restrict__ inIm,
    float* __restrict__ outRe, float* __restrict__ outIm) {
  __shared__ __attribute__((aligned(16))) _Float16 sAre[2][64 * 32];
  __shared__ __attribute__((aligned(16))) _Float16 sAim[2][64 * 32];
  __shared__ __attribute__((aligned(16))) _Float16 sBre[2][64 * 32];
  __shared__ __attribute__((aligned(16))) _Float16 sBim[2][64 * 32];

  const int tid = threadIdx.x;
  const int lane = tid & 31;
  const int wave = tid >> 5;
  const int wm = wave >> 1, wn = wave & 1;
  const int i0 = blockIdx.x * 64;  // M (unitary row) tile
  const int b0 = blockIdx.y * 64;  // N (batch) tile

  v8f cRe[2][2] = {};
  v8f cIm[2][2] = {};
  v4h rAre[4], rAim[4], rBre[4], rBim[4];

  // prologue: stage K-tile 0
  load_regs(Ure, i0, 0, tid, rAre);
  load_regs(Uim, i0, 0, tid, rAim);
  load_regs(inRe, b0, 0, tid, rBre);
  if constexpr (HAS_IM) load_regs(inIm, b0, 0, tid, rBim);
  store_regs(rAre, sAre[0], tid);
  store_regs(rAim, sAim[0], tid);
  store_regs(rBre, sBre[0], tid);
  if constexpr (HAS_IM) store_regs(rBim, sBim[0], tid);
  __syncthreads();

  int cur = 0;
  for (int k0 = 0; k0 < DIM; k0 += 32) {
    const bool nxt = (k0 + 32 < DIM);
    // issue next tile's global loads early; VMEM latency hides under the WMMAs
    if (nxt) {
      load_regs(Ure, i0, k0 + 32, tid, rAre);
      load_regs(Uim, i0, k0 + 32, tid, rAim);
      load_regs(inRe, b0, k0 + 32, tid, rBre);
      if constexpr (HAS_IM) load_regs(inIm, b0, k0 + 32, tid, rBim);
      if (k0 + 64 < DIM) {  // longer-range L2 prefetch (global_prefetch_b8)
        const int pr = tid >> 1;
        const int pc = (tid & 1) * 16;
        __builtin_prefetch(Ure + (size_t)(i0 + pr) * DIM + k0 + 64 + pc, 0, 1);
        __builtin_prefetch(Uim + (size_t)(i0 + pr) * DIM + k0 + 64 + pc, 0, 1);
      }
    }

    v16h ar[2], ai[2], br[2], bi[2], ain[2];
#pragma unroll
    for (int t = 0; t < 2; ++t) {
      ar[t] = load_frag_a(sAre[cur], lane, wm * 32 + t * 16);
      ai[t] = load_frag_a(sAim[cur], lane, wm * 32 + t * 16);
      br[t] = load_frag_b(sBre[cur], lane, wn * 32 + t * 16);
      if constexpr (HAS_IM) {
        bi[t] = load_frag_b(sBim[cur], lane, wn * 32 + t * 16);
        ain[t] = -ai[t];  // f16 WMMA has no A-neg modifier; one xor pass per K-step
      }
    }
#pragma unroll
    for (int tm = 0; tm < 2; ++tm) {
#pragma unroll
      for (int tn = 0; tn < 2; ++tn) {
        cRe[tm][tn] = wmma(ar[tm], br[tn], cRe[tm][tn]);   // Ur*sr
        cIm[tm][tn] = wmma(ai[tm], br[tn], cIm[tm][tn]);   // Ui*sr
        if constexpr (HAS_IM) {
          cRe[tm][tn] = wmma(ain[tm], bi[tn], cRe[tm][tn]);  // -Ui*si
          cIm[tm][tn] = wmma(ar[tm], bi[tn], cIm[tm][tn]);   //  Ur*si
        }
      }
    }

    if (nxt) {
      const int n = cur ^ 1;  // other buffer: nobody reads it this iteration
      store_regs(rAre, sAre[n], tid);
      store_regs(rAim, sAim[n], tid);
      store_regs(rBre, sBre[n], tid);
      if constexpr (HAS_IM) store_regs(rBim, sBim[n], tid);
      __syncthreads();
      cur = n;
    }
  }

  // Epilogue. C/D layout: VGPR r -> M = r + (lane>>4)*8, N = lane&15.
#pragma unroll
  for (int tm = 0; tm < 2; ++tm) {
#pragma unroll
    for (int tn = 0; tn < 2; ++tn) {
      const int m0 = i0 + wm * 32 + tm * 16 + (lane >> 4) * 8;
      const int n = b0 + wn * 32 + tn * 16 + (lane & 15);
      const size_t base = (size_t)n * DIM + m0;
      const v8f re = cRe[tm][tn];
      const v8f im = cIm[tm][tn];
      if constexpr (ABS) {
        const float4 lo = make_float4(sqrtf(re[0] * re[0] + im[0] * im[0]),
                                      sqrtf(re[1] * re[1] + im[1] * im[1]),
                                      sqrtf(re[2] * re[2] + im[2] * im[2]),
                                      sqrtf(re[3] * re[3] + im[3] * im[3]));
        const float4 hi = make_float4(sqrtf(re[4] * re[4] + im[4] * im[4]),
                                      sqrtf(re[5] * re[5] + im[5] * im[5]),
                                      sqrtf(re[6] * re[6] + im[6] * im[6]),
                                      sqrtf(re[7] * re[7] + im[7] * im[7]));
        *(float4*)(outRe + base) = lo;
        *(float4*)(outRe + base + 4) = hi;
      } else {
        *(float4*)(outRe + base) = make_float4(re[0], re[1], re[2], re[3]);
        *(float4*)(outRe + base + 4) = make_float4(re[4], re[5], re[6], re[7]);
        *(float4*)(outIm + base) = make_float4(im[0], im[1], im[2], im[3]);
        *(float4*)(outIm + base + 4) = make_float4(im[4], im[5], im[6], im[7]);
      }
    }
  }
}

// ---------------------------------------------------------------------------
extern "C" void kernel_launch(void* const* d_in, const int* in_sizes, int n_in,
                              void* d_out, int out_size, void* d_ws, size_t ws_size,
                              hipStream_t stream) {
  (void)in_sizes; (void)n_in; (void)out_size; (void)ws_size;
  const float* x = (const float*)d_in[0];
  const float* Ure = (const float*)d_in[1];  // (4, DIM, DIM)
  const float* Uim = (const float*)d_in[2];
  float* out = (float*)d_out;                // (128, 64, 64) == (NB, DIM)

  const size_t SZ = (size_t)NB * DIM;        // 512K floats per state component
  float* Are = (float*)d_ws;                 // ping-pong state buffers (8 MB total)
  float* Aim = Are + SZ;
  float* Bre = Aim + SZ;
  float* Bim = Bre + SZ;
  const size_t US = (size_t)DIM * DIM;

  const dim3 gg(DIM / 64, NB / 64);          // 64 x 2 workgroups per GEMM
  const int permBlocks = (NB * DIM) / 256;

  // encode -> Are (imag = 0, handled by HAS_IM=false in first GEMM)
  encode_kernel<<<NB, 256, 0, stream>>>(x, Are);
  // U[0]  (real input: half the WMMAs)
  cgemm_kernel<false, false><<<gg, 128, 0, stream>>>(Ure, Uim, Are, nullptr, Bre, Bim);
  // CNOT ring
  cnot_kernel<<<permBlocks, 256, 0, stream>>>(Bre, Bim, Are, Aim);
  // U[1]
  cgemm_kernel<true, false><<<gg, 128, 0, stream>>>(Ure + US, Uim + US, Are, Aim, Bre, Bim);
  // QFT
  qft_kernel<<<NB, 256, 0, stream>>>(Bre, Bim, Are, Aim);
  // U[2]
  cgemm_kernel<true, false><<<gg, 128, 0, stream>>>(Ure + 2 * US, Uim + 2 * US, Are, Aim, Bre, Bim);
  // CNOT ring
  cnot_kernel<<<permBlocks, 256, 0, stream>>>(Bre, Bim, Are, Aim);
  // U[3] with fused |.| epilogue straight to d_out
  cgemm_kernel<true, true><<<gg, 128, 0, stream>>>(Ure + 3 * US, Uim + 3 * US, Are, Aim, out, nullptr);
}
